// KANLinear_19713899889300
// MI455X (gfx1250) — compile-verified, matrix operand order checked by reading, not run on hardware
//
#include <hip/hip_runtime.h>
#include <math.h>

// Problem shape (from reference)
static constexpr int NB   = 8192;              // batch
static constexpr int NI   = 1024;              // in features
static constexpr int NO   = 1024;              // out features
static constexpr int KB   = 8;                 // bases per feature (grid 5 + order 3)
static constexpr int KTOT = NI + NI * KB;      // 9216 augmented contraction dim

// GEMM tiling
static constexpr int MT = 128;                 // block M tile
static constexpr int NT = 256;                 // block N tile
static constexpr int KT = 32;                  // bf16 WMMA k
static constexpr int LDSROW = 40;              // padded LDS row stride (elements) = 80B
static constexpr int NK = KTOT / KT;           // 288
static constexpr int ABUF = MT * LDSROW;       // 5120 elems
static constexpr int BBUF = NT * LDSROW;       // 10240 elems
static constexpr int BUFSZ = ABUF + BBUF;      // 15360 elems (30720 B) per buffer

typedef __attribute__((ext_vector_type(16))) __bf16 v16bf;
typedef __attribute__((ext_vector_type(8)))  float  v8f;
typedef unsigned int v4u __attribute__((ext_vector_type(4)));
typedef int          v8i __attribute__((ext_vector_type(8)));
typedef int          v4i __attribute__((ext_vector_type(4)));

__device__ __forceinline__ float knot(int j) { return -4.4f + 0.8f * (float)j; }

// LDS byte offset of a static __shared__ object (generic -> AS(3) -> int)
__device__ __forceinline__ unsigned int lds_addr_of(const void* p) {
    return (unsigned int)(size_t)(__attribute__((address_space(3))) const void*)p;
}

// ---------------------------------------------------------------------------
// Kernel 1: augmented activation matrix A (NB x KTOT) in bf16.
//   A[b, i]            = silu(x[b,i])
//   A[b, NI + i*8 + k] = cubic B-spline basis k of x[b,i]  (Cox-de Boor, EPS)
// ---------------------------------------------------------------------------
__global__ __launch_bounds__(256) void kan_prep_act(const float* __restrict__ x,
                                                    __bf16* __restrict__ A)
{
    int gid = blockIdx.x * 256 + threadIdx.x;       // exactly NB*NI threads
    int b = gid >> 10;
    int i = gid & (NI - 1);
    float xv = x[(size_t)b * NI + i];

    float s = xv * (1.0f / (1.0f + __expf(-xv)));   // silu

    const float EPS = 1e-7f;
    float bb[11];
#pragma unroll
    for (int j = 0; j < 11; ++j)
        bb[j] = (xv >= knot(j) && xv < knot(j + 1)) ? 1.0f : 0.0f;
#pragma unroll
    for (int k = 1; k <= 3; ++k) {
        float inv = 1.0f / ((float)k * 0.8f + EPS);
#pragma unroll
        for (int j = 0; j < 10; ++j)
            bb[j] = (xv - knot(j)) * inv * bb[j] + (knot(j + k + 1) - xv) * inv * bb[j + 1];
    }

    __bf16* rowp = A + (size_t)b * KTOT;
    rowp[i] = (__bf16)s;
    union { __bf16 h8[8]; uint4 u; } pk;
#pragma unroll
    for (int k = 0; k < 8; ++k) pk.h8[k] = (__bf16)bb[k];
    *(uint4*)(rowp + NI + i * 8) = pk.u;
}

// ---------------------------------------------------------------------------
// Kernel 2: augmented weights, TRANSPOSED: Wt (NO x KTOT) bf16, K-contiguous.
//   Wt[j, r]            = base_scaler[r, j]                       (r < NI)
//   Wt[j, NI + i*8 + k] = spline_weight[i, j, k] * spline_scaler[i, j]
// ---------------------------------------------------------------------------
__global__ __launch_bounds__(256) void kan_prep_wt(const float* __restrict__ spline_weight,
                                                   const float* __restrict__ spline_scaler,
                                                   const float* __restrict__ base_scaler,
                                                   __bf16* __restrict__ Wt)
{
    __shared__ __bf16 tile[64][65];
    int r0 = blockIdx.y * 64;
    int j0 = blockIdx.x * 64;
    int t  = threadIdx.x;
#pragma unroll
    for (int l = 0; l < 16; ++l) {
        int idx = t + l * 256;
        int rr = idx >> 6, jj = idx & 63;
        int r = r0 + rr, j = j0 + jj;
        float w;
        if (r < NI) {
            w = base_scaler[(size_t)r * NO + j];
        } else {
            int rs = r - NI;
            int i = rs >> 3, k = rs & 7;
            w = spline_weight[((size_t)i * NO + j) * KB + k] * spline_scaler[(size_t)i * NO + j];
        }
        tile[rr][jj] = (__bf16)w;
    }
    __syncthreads();
#pragma unroll
    for (int l = 0; l < 16; ++l) {
        int idx = t + l * 256;
        int jj = idx >> 6, rr = idx & 63;
        Wt[(size_t)(j0 + jj) * KTOT + (r0 + rr)] = tile[rr][jj];
    }
}

// ---------------------------------------------------------------------------
// TDM descriptor issue: 2D tile (tile_rows x 32 bf16) from a row-major tensor
// (tensor_rows x KTOT, bf16) into LDS with hardware row padding that yields a
// 40-element (80 B) LDS row stride. D# layout per CDNA5 ISA.
//   pad_interval code 3 = every 16 DWORDs (= one 64 B tile row)
//   pad_amount   code 3 = insert 4 DWORDs (= 16 B -> stride 80 B)
// Toolchain here exposes the 6-arg builtin form:
//   (uint32x4 g0, int32x8 g1, int32x4, int32x4, int32x8, i32 cpol)
// ---------------------------------------------------------------------------
#if __has_builtin(__builtin_amdgcn_tensor_load_to_lds)
#define KAN_USE_TDM 1
__device__ __forceinline__ void tdm_load_tile(unsigned int lds_byte_addr,
                                              unsigned long long gaddr,
                                              unsigned int tensor_rows,
                                              unsigned int tile_rows)
{
    v4u g0;
    g0.x = 1u;                                           // count=1, user mode
    g0.y = lds_byte_addr;                                // lds_addr [63:32]
    g0.z = (unsigned int)gaddr;                          // global_addr[31:0]
    g0.w = (unsigned int)(gaddr >> 32) | (2u << 30);     // addr[56:32] | type=2
    v8i g1 = {};
    g1[0] = (int)((1u << 16) | (1u << 20) | (3u << 22) | (3u << 25)); // data_size=2B, pad on
    g1[1] = (int)(((unsigned)KTOT & 0xFFFFu) << 16);     // tensor_dim0 lo16 -> [63:48]
    g1[2] = (int)(((unsigned)KTOT >> 16) | ((tensor_rows & 0xFFFFu) << 16)); // dim0 hi | dim1 lo
    g1[3] = (int)((tensor_rows >> 16) | (32u << 16));    // dim1 hi | tile_dim0=32
    g1[4] = (int)tile_rows;                              // tile_dim1 (tile_dim2=0)
    g1[5] = (int)KTOT;                                   // tensor_dim0_stride lo32
    g1[6] = 0;                                           // stride hi | dim1_stride lo
    g1[7] = 0;
    v4i z4 = {0, 0, 0, 0};
    v8i z8 = {};
    __builtin_amdgcn_tensor_load_to_lds(g0, g1, z4, z4, z8, 0);
}
#else
#define KAN_USE_TDM 0
#endif

// ---------------------------------------------------------------------------
// Kernel 3: C(NB x NO) = A(NB x KTOT) * Wt(NO x KTOT)^T, bf16 -> f32 WMMA.
// Block tile 128x256, 8 wave32s as 2(M) x 4(N), wave tile 64x64 = 4x4 WMMA.
// Double-buffered LDS filled by the Tensor Data Mover (wave 0 issues,
// TENSORcnt-waited), 16 v_wmma per wave per K-step.
// ---------------------------------------------------------------------------
__global__ __launch_bounds__(256) void kan_gemm(const __bf16* __restrict__ A,
                                                const __bf16* __restrict__ Wt,
                                                float* __restrict__ out)
{
    __shared__ __align__(16) __bf16 smem[2 * BUFSZ];   // 61440 B

    const int t    = threadIdx.x;
    const int wave = t >> 5;
    const int lane = t & 31;
    const int m0 = blockIdx.y * MT;
    const int n0 = blockIdx.x * NT;
    const int wm = (wave & 1) * 64;        // wave M offset within block tile
    const int wn = (wave >> 1) * 64;       // wave N offset within block tile

    v8f acc[4][4];
#pragma unroll
    for (int mi = 0; mi < 4; ++mi)
#pragma unroll
        for (int ni = 0; ni < 4; ++ni)
#pragma unroll
            for (int v = 0; v < 8; ++v) acc[mi][ni][v] = 0.0f;

    const unsigned long long aTile0 =
        (unsigned long long)(size_t)(A  + (size_t)m0 * KTOT);
    const unsigned long long bTile0 =
        (unsigned long long)(size_t)(Wt + (size_t)n0 * KTOT);

    // ISA 16-bit fragment lane mapping: lanes 0-15 K{0-7,16-23}, 16-31 K{8-15,24-31}
    const int lrow  = lane & 15;
    const int khalf = (lane >> 4) * 8;

#if KAN_USE_TDM
    const unsigned int ldsBase = lds_addr_of(smem);

    if (wave == 0) {   // preload K-tile 0 into buffer 0 via TDM
        tdm_load_tile(ldsBase,                 aTile0, (unsigned)NB, (unsigned)MT);
        tdm_load_tile(ldsBase + ABUF * 2,      bTile0, (unsigned)NO, (unsigned)NT);
        __builtin_amdgcn_s_wait_tensorcnt((short)0);
    }
    __syncthreads();

    int buf = 0;
    for (int kt = 0; kt < NK; ++kt) {
        const bool more = (kt + 1 < NK);
        if (more && wave == 0) {           // async prefetch of next K-tile
            unsigned int nbase = ldsBase + (unsigned)((buf ^ 1) * BUFSZ * 2);
            unsigned long long koff = (unsigned long long)(kt + 1) * (KT * 2);
            tdm_load_tile(nbase,            aTile0 + koff, (unsigned)NB, (unsigned)MT);
            tdm_load_tile(nbase + ABUF * 2, bTile0 + koff, (unsigned)NO, (unsigned)NT);
        }

        const __bf16* Ab = smem + buf * BUFSZ;
        const __bf16* Bb = Ab + ABUF;
        union { v16bf v; uint4 u[2]; } fa[4], fb[4];
#pragma unroll
        for (int mi = 0; mi < 4; ++mi) {
            const __bf16* p = Ab + (wm + mi * 16 + lrow) * LDSROW;
            fa[mi].u[0] = *(const uint4*)(p + khalf);
            fa[mi].u[1] = *(const uint4*)(p + 16 + khalf);
        }
#pragma unroll
        for (int ni = 0; ni < 4; ++ni) {
            const __bf16* p = Bb + (wn + ni * 16 + lrow) * LDSROW;
            fb[ni].u[0] = *(const uint4*)(p + khalf);
            fb[ni].u[1] = *(const uint4*)(p + 16 + khalf);
        }
#pragma unroll
        for (int mi = 0; mi < 4; ++mi)
#pragma unroll
            for (int ni = 0; ni < 4; ++ni)
                acc[mi][ni] = __builtin_amdgcn_wmma_f32_16x16x32_bf16(
                    false, fa[mi].v, false, fb[ni].v,
                    (short)0, acc[mi][ni], false, false);

        if (more && wave == 0)
            __builtin_amdgcn_s_wait_tensorcnt((short)0);  // tile kt+1 resident
        __syncthreads();
        buf ^= 1;
    }
#else
    // Fallback: synchronous register staging (no TDM builtin on this toolchain)
    const __bf16* aB = A  + (size_t)m0 * KTOT;
    const __bf16* bB = Wt + (size_t)n0 * KTOT;
    for (int kt = 0; kt < NK; ++kt) {
        __syncthreads();
#pragma unroll
        for (int c = 0; c < 6; ++c) {          // (128+256)*32/8 = 1536 chunks
            int idx = t + c * 256;
            const __bf16* src; __bf16* dst;
            if (idx < MT * 4) {
                int row = idx >> 2, q = (idx & 3) * 8;
                src = aB + (size_t)row * KTOT + kt * KT + q;
                dst = smem + row * LDSROW + q;
            } else {
                int idx2 = idx - MT * 4;
                int row = idx2 >> 2, q = (idx2 & 3) * 8;
                src = bB + (size_t)row * KTOT + kt * KT + q;
                dst = smem + ABUF + row * LDSROW + q;
            }
            *(uint4*)dst = *(const uint4*)src;
        }
        __syncthreads();
        union { v16bf v; uint4 u[2]; } fa[4], fb[4];
#pragma unroll
        for (int mi = 0; mi < 4; ++mi) {
            const __bf16* p = smem + (wm + mi * 16 + lrow) * LDSROW;
            fa[mi].u[0] = *(const uint4*)(p + khalf);
            fa[mi].u[1] = *(const uint4*)(p + 16 + khalf);
        }
#pragma unroll
        for (int ni = 0; ni < 4; ++ni) {
            const __bf16* p = smem + ABUF + (wn + ni * 16 + lrow) * LDSROW;
            fb[ni].u[0] = *(const uint4*)(p + khalf);
            fb[ni].u[1] = *(const uint4*)(p + 16 + khalf);
        }
#pragma unroll
        for (int mi = 0; mi < 4; ++mi)
#pragma unroll
            for (int ni = 0; ni < 4; ++ni)
                acc[mi][ni] = __builtin_amdgcn_wmma_f32_16x16x32_bf16(
                    false, fa[mi].v, false, fb[ni].v,
                    (short)0, acc[mi][ni], false, false);
    }
#endif

    // C/D layout (ISA 7.12.2): VGPR v -> M = v + 8*(lane>=16), N = lane&15
    const int orow = (lane >> 4) * 8;
    const int ocol = lane & 15;
#pragma unroll
    for (int mi = 0; mi < 4; ++mi)
#pragma unroll
        for (int ni = 0; ni < 4; ++ni) {
            float* po = out + (size_t)(m0 + wm + mi * 16 + orow) * NO
                            + (n0 + wn + ni * 16 + ocol);
#pragma unroll
            for (int v = 0; v < 8; ++v)
                po[(size_t)v * NO] = acc[mi][ni][v];
        }
}

// ---------------------------------------------------------------------------
extern "C" void kernel_launch(void* const* d_in, const int* in_sizes, int n_in,
                              void* d_out, int out_size, void* d_ws, size_t ws_size,
                              hipStream_t stream)
{
    (void)in_sizes; (void)n_in; (void)out_size; (void)ws_size;
    const float* x  = (const float*)d_in[0];   // (8192, 1024)
    const float* sw = (const float*)d_in[1];   // (1024, 1024, 8)
    const float* ss = (const float*)d_in[2];   // (1024, 1024)
    const float* bs = (const float*)d_in[3];   // (1024, 1024)
    float* out = (float*)d_out;                // (8192, 1024) f32

    // workspace: A bf16 (NB x KTOT) = 151 MB, then Wt bf16 (NO x KTOT) = 19 MB
    __bf16* A  = (__bf16*)d_ws;
    __bf16* Wt = (__bf16*)((char*)d_ws + (size_t)NB * KTOT * 2);

    kan_prep_act<<<(NB * NI) / 256, 256, 0, stream>>>(x, A);
    kan_prep_wt<<<dim3(NO / 64, KTOT / 64), 256, 0, stream>>>(sw, ss, bs, Wt);
    kan_gemm<<<dim3(NO / NT, NB / MT), 256, 0, stream>>>(A, Wt, out);
}
